// EfficientVitBlock_50637664420537
// MI455X (gfx1250) — matrix operand order repaired and checked by reading, not compiled, not run on hardware
//
#include <hip/hip_runtime.h>
#include <hip/hip_bf16.h>

typedef __attribute__((ext_vector_type(16))) _Float16 v16h;
typedef __attribute__((ext_vector_type(8)))  _Float16 v8h;
typedef __attribute__((ext_vector_type(4)))  _Float16 v4h;
typedef __attribute__((ext_vector_type(2)))  _Float16 v2h;
typedef __attribute__((ext_vector_type(8)))  float    v8f;
typedef __attribute__((ext_vector_type(4)))  float    v4f;

#define BN   32
#define CDIM 256
#define HCH  512
#define IMG  56
#define HW   3136
#define WSZ  7
#define NWIN 64
#define NPIX 49
#define QSCALE 0.25f   // KD=16 -> 16^-0.5

// ---------------------------------------------------------------- utilities
__global__ void convert_f32_f16(const float* __restrict__ in,
                                _Float16* __restrict__ out, int n) {
  int i = blockIdx.x * blockDim.x + threadIdx.x;
  if (i < n) out[i] = (_Float16)in[i];
}

// ---------------------------------------- depthwise 3x3 + BN + residual add
__global__ void dw3x3_res(const float* __restrict__ x, const float* __restrict__ w,
                          const float* __restrict__ s, const float* __restrict__ b,
                          float* __restrict__ out) {
  int idx = blockIdx.x * blockDim.x + threadIdx.x;
  if (idx >= BN * CDIM * HW) return;
  int p = idx % HW;
  int c = (idx / HW) % CDIM;
  int n = idx / (HW * CDIM);
  int h = p / IMG, wc = p % IMG;
  const float* xc = x + ((size_t)n * CDIM + c) * HW;
  const float* wk = w + c * 9;
  float acc = 0.f;
#pragma unroll
  for (int i = 0; i < 3; ++i) {
    int hh = h + i - 1;
    if (hh < 0 || hh >= IMG) continue;
#pragma unroll
    for (int j = 0; j < 3; ++j) {
      int ww = wc + j - 1;
      if (ww < 0 || ww >= IMG) continue;
      acc += wk[i * 3 + j] * xc[hh * IMG + ww];
    }
  }
  out[idx] = x[idx] + s[c] * acc + b[c];
}

// ------------------------- window-local depthwise 5x5 (+opt BN) * postmul
__global__ void win_dw5x5(const float* __restrict__ src, int chTot, int nch,
                          const float* __restrict__ w5, const float* __restrict__ s,
                          const float* __restrict__ b, float postmul,
                          float* __restrict__ dst) {
  int idx = blockIdx.x * blockDim.x + threadIdx.x;
  if (idx >= BN * nch * HW) return;
  int p = idx % HW;
  int c = (idx / HW) % nch;
  int n = idx / (HW * nch);
  int h = p / IMG, wc = p % IMG;
  int li = h % WSZ, lj = wc % WSZ;
  int h0 = h - li, w0 = wc - lj;
  const float* xc = src + ((size_t)n * chTot + c) * HW;
  const float* wk = w5 + c * 25;
  float acc = 0.f;
#pragma unroll
  for (int i = 0; i < 5; ++i) {
    int gi = li + i - 2;
    if (gi < 0 || gi >= WSZ) continue;
#pragma unroll
    for (int j = 0; j < 5; ++j) {
      int gj = lj + j - 2;
      if (gj < 0 || gj >= WSZ) continue;
      acc += wk[i * 5 + j] * xc[(h0 + gi) * IMG + (w0 + gj)];
    }
  }
  float v = (s ? acc * s[c] : acc) + b[c];
  dst[((size_t)n * nch + c) * HW + p] = v * postmul;
}

// ----------------------------------------------------- WMMA tiled GEMM
// out[n][m][p] = epi( sum_k A[m][k] * B[n][k][p] )
// 256 threads = 8 waves; block tile 128(M) x 64(N); wave tile 16x64; K step 32.
// LDS: A row-major (lane frag = 2 contiguous v8h runs), B transposed n-major
// (lane frag = 2 contiguous v8h runs) => all fragment reads are ds_load_b128.
// K-pair packed transpose stores => ds_store_b32. 32-bit offsets throughout.
template <bool B_F16, bool RELU_OUT, bool RESIDUAL, bool OUT_F16>
__global__ __launch_bounds__(256)
void gemm_wmma(const _Float16* __restrict__ A, const void* __restrict__ Bsrc,
               const float* __restrict__ scale, const float* __restrict__ bias,
               const float* __restrict__ res, void* __restrict__ out,
               int M, int K) {
  __shared__ alignas(16) _Float16 lA[128][40];  // 128 M rows x 32 K (+pad, 80B stride)
  __shared__ alignas(16) _Float16 lBt[64][40];  // 64 N rows x 32 K (+pad)
  const int m0  = blockIdx.x * 128;
  const int n0  = blockIdx.y * 64;
  const int img = blockIdx.z;
  // per-image 32-bit-addressable bases
  const float*    Bf = (const float*)Bsrc + (size_t)img * K * HW;
  const _Float16* Bh = (const _Float16*)Bsrc + (size_t)img * K * HW;
  const float*    rimg = RESIDUAL ? res + (size_t)img * M * HW : nullptr;
  float*    of = (float*)out + (OUT_F16 ? 0 : (size_t)img * M * HW);
  _Float16* oh = (_Float16*)out + (OUT_F16 ? (size_t)img * M * HW : 0);

  const int tid  = threadIdx.x;
  const int lane = tid & 31;
  const int wave = tid >> 5;
  const int hf   = lane >> 4;
  const int l16  = lane & 15;

  v8f acc[4] = {};

  for (int k0 = 0; k0 < K; k0 += 32) {
    // --- A tile 128x32: 512 v8h chunks, global b128 -> LDS b128
#pragma unroll
    for (int r = 0; r < 2; ++r) {
      unsigned t = tid + r * 256;
      unsigned mr = t >> 2, ch = (t & 3) * 8;
      *(v8h*)&lA[mr][ch] = *(const v8h*)(A + (unsigned)(m0 + mr) * K + k0 + ch);
    }
    // --- B tile 32(K)x64(N) -> transposed LDS, K-pairs packed into b32 stores
    {
      unsigned kp = (tid >> 4) * 2;          // even K row of the pair
      unsigned nc = (tid & 15) * 4;          // 4 consecutive N columns
      if (B_F16) {
        v4h a = *(const v4h*)(Bh + (unsigned)(k0 + kp) * HW + n0 + nc);
        v4h c = *(const v4h*)(Bh + (unsigned)(k0 + kp + 1) * HW + n0 + nc);
#pragma unroll
        for (int j = 0; j < 4; ++j) {
          v2h h2 = {a[j], c[j]};
          *(v2h*)&lBt[nc + j][kp] = h2;
        }
      } else {
        v4f a = *(const v4f*)(Bf + (unsigned)(k0 + kp) * HW + n0 + nc);
        v4f c = *(const v4f*)(Bf + (unsigned)(k0 + kp + 1) * HW + n0 + nc);
#pragma unroll
        for (int j = 0; j < 4; ++j) {
          v2h h2 = {(_Float16)a[j], (_Float16)c[j]};
          *(v2h*)&lBt[nc + j][kp] = h2;
        }
      }
    }
    if (k0 + 32 < K) {  // prefetch next B tile row
      unsigned gi = (unsigned)(k0 + 32 + lane) * HW + n0;
      __builtin_prefetch(B_F16 ? (const void*)(Bh + gi) : (const void*)(Bf + gi),
                         0, 1);
    }
    __syncthreads();

    // A fragment: two aligned 16B runs of row (wave*16+l16)
    const int arow = wave * 16 + l16;
    v8h alo = *(const v8h*)&lA[arow][8 * hf];
    v8h ahi = *(const v8h*)&lA[arow][16 + 8 * hf];
    v16h a = __builtin_shufflevector(alo, ahi, 0, 1, 2, 3, 4, 5, 6, 7,
                                     8, 9, 10, 11, 12, 13, 14, 15);
#pragma unroll
    for (int t = 0; t < 4; ++t) {
      const int brow = t * 16 + l16;
      v8h blo = *(const v8h*)&lBt[brow][8 * hf];
      v8h bhi = *(const v8h*)&lBt[brow][16 + 8 * hf];
      v16h b = __builtin_shufflevector(blo, bhi, 0, 1, 2, 3, 4, 5, 6, 7,
                                       8, 9, 10, 11, 12, 13, 14, 15);
      acc[t] = __builtin_amdgcn_wmma_f32_16x16x32_f16(
          false, a, false, b, (short)0, acc[t], false, false);
    }
    __syncthreads();
  }

  // epilogue: scale/bias (+residual, +relu), f32 or f16 store (32-bit offsets)
#pragma unroll
  for (int t = 0; t < 4; ++t) {
#pragma unroll
    for (int r = 0; r < 8; ++r) {
      unsigned m = m0 + wave * 16 + 8 * hf + r;
      unsigned n = n0 + t * 16 + l16;
      unsigned oi = m * (unsigned)HW + n;
      float v = acc[t][r] * scale[m] + bias[m];
      if (RESIDUAL) v += rimg[oi];
      if (RELU_OUT) v = fmaxf(v, 0.f);
      if (OUT_F16) oh[oi] = (_Float16)v;
      else         of[oi] = v;
    }
  }
}

// -------------------------------------------- fused per-window attention
__global__ __launch_bounds__(256)
void win_attention(const float* __restrict__ Q,      // [n][64][HW] (already *0.25)
                   const float* __restrict__ FEAT,   // [n][128][HW]; k = ch 64..127
                   const float* __restrict__ V,      // [n][256][HW]
                   const float* __restrict__ biases, // attn_biases row 0 (49 vals)
                   float* __restrict__ O) {          // [n][256][HW]
  __shared__ float smem[CDIM * NPIX];  // q|k first, reused for v
  __shared__ float la[NPIX][NPIX];
  const int n = blockIdx.y;
  const int win = blockIdx.x;
  const int hb = (win / 8) * WSZ, wb = (win % 8) * WSZ;
  const int tid = threadIdx.x;

  for (int t = tid; t < 64 * NPIX; t += 256) {
    int c = t / NPIX, p = t % NPIX;
    int g = (hb + p / WSZ) * IMG + (wb + p % WSZ);
    smem[t]             = Q[((size_t)n * 64 + c) * HW + g];
    smem[64 * NPIX + t] = FEAT[((size_t)n * 128 + 64 + c) * HW + g];
  }
  __syncthreads();

  const float* lq = smem;
  const float* lk = smem + 64 * NPIX;
  for (int t = tid; t < NPIX * NPIX; t += 256) {
    int i = t / NPIX, j = t % NPIX;
    float acc = 0.f;
#pragma unroll 8
    for (int c = 0; c < 64; ++c) acc += lq[c * NPIX + i] * lk[c * NPIX + j];
    int d = abs(i / WSZ - j / WSZ) * WSZ + abs(i % WSZ - j % WSZ);
    la[i][j] = acc + biases[d];
  }
  __syncthreads();

  if (tid < NPIX) {  // softmax row tid
    float m = -1e30f;
    for (int j = 0; j < NPIX; ++j) m = fmaxf(m, la[tid][j]);
    float sum = 0.f;
    for (int j = 0; j < NPIX; ++j) {
      float e = __expf(la[tid][j] - m);
      la[tid][j] = e;
      sum += e;
    }
    float inv = 1.f / sum;
    for (int j = 0; j < NPIX; ++j) la[tid][j] *= inv;
  }
  __syncthreads();

  for (int t = tid; t < CDIM * NPIX; t += 256) {  // v overwrites q/k
    int c = t / NPIX, p = t % NPIX;
    int g = (hb + p / WSZ) * IMG + (wb + p % WSZ);
    smem[t] = V[((size_t)n * CDIM + c) * HW + g];
  }
  __syncthreads();

  for (int t = tid; t < CDIM * NPIX; t += 256) {
    int c = t / NPIX, i = t % NPIX;
    float acc = 0.f;
#pragma unroll 7
    for (int j = 0; j < NPIX; ++j) acc += smem[c * NPIX + j] * la[i][j];
    int g = (hb + i / WSZ) * IMG + (wb + i % WSZ);
    O[((size_t)n * CDIM + c) * HW + g] = fmaxf(acc, 0.f);  // relu fused (pre-proj)
  }
}

// ---------------------------------------------------------------- driver
extern "C" void kernel_launch(void* const* d_in, const int* in_sizes, int n_in,
                              void* d_out, int out_size, void* d_ws, size_t ws_size,
                              hipStream_t stream) {
  const float* x       = (const float*)d_in[0];
  const float* dw0_w   = (const float*)d_in[1];
  const float* dw0_s   = (const float*)d_in[2];
  const float* dw0_b   = (const float*)d_in[3];
  const float* ffn0_w1 = (const float*)d_in[4];
  const float* ffn0_s1 = (const float*)d_in[5];
  const float* ffn0_b1 = (const float*)d_in[6];
  const float* ffn0_w2 = (const float*)d_in[7];
  const float* ffn0_s2 = (const float*)d_in[8];
  const float* ffn0_b2 = (const float*)d_in[9];
  const float* qkv_w   = (const float*)d_in[10];
  const float* qkv_s   = (const float*)d_in[11];
  const float* qkv_b   = (const float*)d_in[12];
  const float* dws_w   = (const float*)d_in[13];
  const float* dws_s   = (const float*)d_in[14];
  const float* dws_b   = (const float*)d_in[15];
  const float* mix_w   = (const float*)d_in[16];
  const float* mix_b   = (const float*)d_in[17];
  const float* attn_bi = (const float*)d_in[18];
  const float* proj_w  = (const float*)d_in[19];
  const float* proj_s  = (const float*)d_in[20];
  const float* proj_b  = (const float*)d_in[21];
  const float* dw1_w   = (const float*)d_in[22];
  const float* dw1_s   = (const float*)d_in[23];
  const float* dw1_b   = (const float*)d_in[24];
  const float* ffn1_w1 = (const float*)d_in[25];
  const float* ffn1_s1 = (const float*)d_in[26];
  const float* ffn1_b1 = (const float*)d_in[27];
  const float* ffn1_w2 = (const float*)d_in[28];
  const float* ffn1_s2 = (const float*)d_in[29];
  const float* ffn1_b2 = (const float*)d_in[30];

  const size_t S = (size_t)BN * CDIM * HW * sizeof(float);  // 102,760,448 B
  char* ws = (char*)d_ws;
  float*    XA   = (float*)ws;                    // dw0 out; later V; later XD
  _Float16* H16  = (_Float16*)(ws + S);           // ffn hidden f16; later O
  float*    FEAT = (float*)(ws + 2 * S);          // qkv out (128 ch)
  float*    Qb   = (float*)(ws + 2 * S + S / 2);  // q after dws (64 ch)
  float*    Vb   = XA;
  float*    Ob   = (float*)(ws + S);
  float*    XD   = XA;
  _Float16* WF   = (_Float16*)(ws + 2 * S + S / 2 + S / 4);
  _Float16* w1h   = WF;
  _Float16* w2h   = w1h + (size_t)HCH * CDIM;
  _Float16* qkvh  = w2h + (size_t)CDIM * HCH;
  _Float16* projh = qkvh + (size_t)128 * CDIM;
  _Float16* w1bh  = projh + (size_t)CDIM * CDIM;
  _Float16* w2bh  = w1bh + (size_t)HCH * CDIM;
  float* XB = (float*)d_out;  // post-ffn0 activation lives in d_out until ffn1

  auto cvt = [&](const float* src, _Float16* dst, int n) {
    convert_f32_f16<<<(n + 255) / 256, 256, 0, stream>>>(src, dst, n);
  };
  cvt(ffn0_w1, w1h, HCH * CDIM);
  cvt(ffn0_w2, w2h, CDIM * HCH);
  cvt(qkv_w, qkvh, 128 * CDIM);
  cvt(proj_w, projh, CDIM * CDIM);
  cvt(ffn1_w1, w1bh, HCH * CDIM);
  cvt(ffn1_w2, w2bh, CDIM * HCH);

  const int tot = BN * CDIM * HW;
  dw3x3_res<<<(tot + 255) / 256, 256, 0, stream>>>(x, dw0_w, dw0_s, dw0_b, XA);

  dim3 g512(HCH / 128, HW / 64, BN);
  dim3 g256(CDIM / 128, HW / 64, BN);
  dim3 g128(128 / 128, HW / 64, BN);
  // ffn0: hidden = relu(bn(W1 @ XA)) stored f16
  gemm_wmma<false, true, false, true><<<g512, 256, 0, stream>>>(
      w1h, XA, ffn0_s1, ffn0_b1, nullptr, H16, HCH, CDIM);
  // ffn0: XB = XA + bn(W2 @ hidden)
  gemm_wmma<true, false, true, false><<<g256, 256, 0, stream>>>(
      w2h, H16, ffn0_s2, ffn0_b2, XA, XB, CDIM, HCH);
  // qkv (pointwise => full image): FEAT = bn(Wqkv @ XB)
  gemm_wmma<false, false, false, false><<<g128, 256, 0, stream>>>(
      qkvh, XB, qkv_s, qkv_b, nullptr, FEAT, 128, CDIM);
  // q = dws5x5(q) * 0.25 ; v = mix5x5(XB) + b   (window-local padding)
  win_dw5x5<<<(BN * 64 * HW + 255) / 256, 256, 0, stream>>>(
      FEAT, 128, 64, dws_w, dws_s, dws_b, QSCALE, Qb);
  win_dw5x5<<<(tot + 255) / 256, 256, 0, stream>>>(
      XB, CDIM, CDIM, mix_w, nullptr, mix_b, 1.f, Vb);
  // fused window attention -> relu(O)
  dim3 ga(NWIN, BN);
  win_attention<<<ga, 256, 0, stream>>>(Qb, FEAT, Vb, attn_bi, Ob);
  // proj with in-place residual on XB (=d_out)
  gemm_wmma<false, false, true, false><<<g256, 256, 0, stream>>>(
      projh, Ob, proj_s, proj_b, XB, XB, CDIM, CDIM);
  // dw1 + residual
  dw3x3_res<<<(tot + 255) / 256, 256, 0, stream>>>(XB, dw1_w, dw1_s, dw1_b, XD);
  // ffn1
  gemm_wmma<false, true, false, true><<<g512, 256, 0, stream>>>(
      w1bh, XD, ffn1_s1, ffn1_b1, nullptr, H16, HCH, CDIM);
  gemm_wmma<true, false, true, false><<<g256, 256, 0, stream>>>(
      w2bh, H16, ffn1_s2, ffn1_b2, XD, (float*)d_out, CDIM, HCH);
}
